// HebbianLayer_1348619731086
// MI455X (gfx1250) — matrix operand order repaired and checked by reading, not compiled, not run on hardware
//
#include <hip/hip_runtime.h>

// ---------------------------------------------------------------------------
// Types / helpers
// ---------------------------------------------------------------------------
typedef __attribute__((ext_vector_type(16))) __bf16       v16bf;
typedef __attribute__((ext_vector_type(8)))  float        v8f;
typedef __attribute__((ext_vector_type(4)))  unsigned int u32x4;

union Frag {
  u32x4          u[2];
  unsigned short s[16];
  v16bf          v;
};

__device__ __forceinline__ unsigned short f2bf(float f) {
  unsigned u = __builtin_bit_cast(unsigned, f);
  unsigned r = (u + 0x7FFFu + ((u >> 16) & 1u)) >> 16;   // RNE
  return (unsigned short)r;
}
__device__ __forceinline__ float bf2f(unsigned short h) {
  unsigned u = ((unsigned)h) << 16;
  return __builtin_bit_cast(float, u);
}
__device__ __forceinline__ v8f v8zero() {
  v8f z = {0.f, 0.f, 0.f, 0.f, 0.f, 0.f, 0.f, 0.f};
  return z;
}
__device__ __forceinline__ v8f wmma_bf16(v16bf a, v16bf b, v8f c) {
  return __builtin_amdgcn_wmma_f32_16x16x32_bf16(false, a, false, b, (short)0, c,
                                                 false, false);
}
__device__ __forceinline__ float silu(float x) { return x / (1.f + expf(-x)); }

// Problem constants (match reference setup_inputs)
constexpr int B_  = 8;
constexpr int T_  = 4096;
constexpr int D_  = 768;
constexpr int DI_ = 1536;
constexpr int BT_ = B_ * T_;
constexpr int C_  = 64;    // CHUNK
constexpr int NC_ = T_ / C_;

// ---------------------------------------------------------------------------
// K0: fp32 -> bf16 cast (weights)
// ---------------------------------------------------------------------------
__global__ void cast_kernel(const float* __restrict__ src,
                            unsigned short* __restrict__ dst, int n) {
  int i = blockIdx.x * 256 + threadIdx.x;
  if (i < n) dst[i] = f2bf(src[i]);
}

// ---------------------------------------------------------------------------
// K1: RMSNorm + bf16 cast.  One 256-thread block per row of D_=768.
// ---------------------------------------------------------------------------
__global__ void rmsnorm_kernel(const float* __restrict__ x,
                               const float* __restrict__ w,
                               unsigned short* __restrict__ xn) {
  __shared__ float red[8];
  size_t row = blockIdx.x;
  const float* xr = x + row * (size_t)D_;
  int tid = threadIdx.x;
  float s = 0.f;
  for (int i = tid; i < D_; i += 256) { float v = xr[i]; s += v * v; }
  for (int o = 16; o > 0; o >>= 1) s += __shfl_xor(s, o, 32);
  if ((tid & 31) == 0) red[tid >> 5] = s;
  __syncthreads();
  if (tid == 0) {
    float t = 0.f;
    for (int i = 0; i < 8; i++) t += red[i];
    red[0] = rsqrtf(t / (float)D_ + 1e-5f);
  }
  __syncthreads();
  float sc = red[0];
  unsigned short* orow = xn + row * (size_t)D_;
  for (int i = tid; i < D_; i += 256) orow[i] = f2bf(xr[i] * sc * w[i]);
}

// ---------------------------------------------------------------------------
// WMMA fragment loaders (global, bf16 / f32 source)
//   A fragment (16x32, MxK): lane m = lane&15, g = lane>>4
//     elements 0..7  -> K = k0 + 8g .. +7     (16B contiguous)
//     elements 8..15 -> K = k0 + 16 + 8g ..   (16B contiguous)
//   B fragment (32x16, KxN) from row-major W[N,K]: lane n = lane&15
//     elements 0..7 -> K = k0 + 8g .., 8..15 -> K = k0 + 16 + 8g ..
// ---------------------------------------------------------------------------
__device__ __forceinline__ void load_a16(Frag& f, const unsigned short* row,
                                         int k0, int g) {
  f.u[0] = *(const u32x4*)(row + k0 + 8 * g);
  f.u[1] = *(const u32x4*)(row + k0 + 16 + 8 * g);
}
__device__ __forceinline__ void load_a32(Frag& f, const float* row, int k0, int g) {
  const float* p0 = row + k0 + 8 * g;
  const float* p1 = row + k0 + 16 + 8 * g;
#pragma unroll
  for (int i = 0; i < 8; i++) { f.s[i] = f2bf(p0[i]); f.s[8 + i] = f2bf(p1[i]); }
}
__device__ __forceinline__ void load_b16(Frag& f, const unsigned short* row,
                                         int k0, int g) {
  f.u[0] = *(const u32x4*)(row + k0 + 8 * g);
  f.u[1] = *(const u32x4*)(row + k0 + 16 + 8 * g);
}

// ---------------------------------------------------------------------------
// K2: tiled WMMA GEMM  C[M,N] = A[M,K] * W[N,K]^T, block tile 128x64,
// 8 waves (4x2), each wave a 32x32 tile (2x2 accumulators).
// EPI: 0 = store bf16, 1 = silu->bf16, 2 = f32 + bf16, 3 = x+out+alpha*acc->f32
// ---------------------------------------------------------------------------
template <int EPI, bool AF32>
__global__ void gemm_kernel(const void* __restrict__ Ap,
                            const unsigned short* __restrict__ W, int M, int N,
                            int K, float* __restrict__ Cf,
                            unsigned short* __restrict__ Cbf,
                            const float* __restrict__ xres,
                            const float* __restrict__ outres,
                            const float* __restrict__ logalpha) {
  int tid = threadIdx.x;
  int w = tid >> 5, lane = tid & 31, g = lane >> 4, hn = lane & 15;
  int wm = w & 3, wn = w >> 2;
  size_t m0 = (size_t)blockIdx.x * 128 + (size_t)wm * 32;
  size_t n0 = (size_t)blockIdx.y * 64 + (size_t)wn * 32;

  v8f c00 = v8zero(), c01 = v8zero(), c10 = v8zero(), c11 = v8zero();

  const unsigned short* a16r0 = nullptr;
  const unsigned short* a16r1 = nullptr;
  const float* a32r0 = nullptr;
  const float* a32r1 = nullptr;
  if (AF32) {
    const float* A = (const float*)Ap;
    a32r0 = A + (m0 + hn) * (size_t)K;
    a32r1 = A + (m0 + 16 + hn) * (size_t)K;
  } else {
    const unsigned short* A = (const unsigned short*)Ap;
    a16r0 = A + (m0 + hn) * (size_t)K;
    a16r1 = A + (m0 + 16 + hn) * (size_t)K;
  }
  const unsigned short* br0 = W + (n0 + hn) * (size_t)K;
  const unsigned short* br1 = W + (n0 + 16 + hn) * (size_t)K;

  for (int k0 = 0; k0 < K; k0 += 32) {
    Frag a0, a1, b0, b1;
    if (AF32) {
      load_a32(a0, a32r0, k0, g);
      load_a32(a1, a32r1, k0, g);
    } else {
      load_a16(a0, a16r0, k0, g);
      load_a16(a1, a16r1, k0, g);
      if (k0 + 256 < K) {                       // global_prefetch_b8 path
        __builtin_prefetch(a16r0 + k0 + 256, 0, 1);
        __builtin_prefetch(a16r1 + k0 + 256, 0, 1);
      }
    }
    load_b16(b0, br0, k0, g);
    load_b16(b1, br1, k0, g);
    c00 = wmma_bf16(a0.v, b0.v, c00);
    c01 = wmma_bf16(a0.v, b1.v, c01);
    c10 = wmma_bf16(a1.v, b0.v, c10);
    c11 = wmma_bf16(a1.v, b1.v, c11);
  }

  float alpha = 0.f;
  if (EPI == 3) alpha = expf(logalpha[0]);

#pragma unroll
  for (int t = 0; t < 4; t++) {
    v8f acc = (t == 0) ? c00 : (t == 1) ? c01 : (t == 2) ? c10 : c11;
    size_t mb = m0 + ((t >> 1) ? 16 : 0);
    size_t nb = n0 + ((t & 1) ? 16 : 0);
#pragma unroll
    for (int v = 0; v < 8; v++) {
      size_t m = mb + v + 8 * g;
      size_t idx = m * (size_t)N + nb + hn;
      float val = acc[v];
      if (EPI == 0) {
        Cbf[idx] = f2bf(val);
      } else if (EPI == 1) {
        Cbf[idx] = f2bf(silu(val));
      } else if (EPI == 2) {
        Cf[idx] = val;
        Cbf[idx] = f2bf(val);
      } else {
        Cf[idx] = xres[idx] + outres[idx] + alpha * val;
      }
    }
  }
}

// ---------------------------------------------------------------------------
// K3: depthwise causal conv (width 4) + bias + silu, * silu-gate (in-place)
// ---------------------------------------------------------------------------
__global__ void conv_silu_gate_kernel(const unsigned short* __restrict__ val,
                                      unsigned short* __restrict__ gate_act,
                                      const float* __restrict__ conv_w,
                                      const float* __restrict__ conv_b) {
  long idx = (long)blockIdx.x * 256 + threadIdx.x;  // over BT_*DI_
  int e = (int)(idx % DI_);
  long bl = idx / DI_;
  int l = (int)(bl % T_);
  long bb = bl / T_;
  float acc = conv_b[e];
#pragma unroll
  for (int j = 0; j < 4; j++) {
    int lt = l - 3 + j;
    if (lt >= 0)
      acc += conv_w[e * 4 + j] * bf2f(val[(bb * T_ + lt) * (long)DI_ + e]);
  }
  float g = bf2f(gate_act[idx]);
  gate_act[idx] = f2bf(silu(acc) * g);
}

// ---------------------------------------------------------------------------
// K4: intra-chunk attention (scan-independent part).
//   One 128-thread (4-wave) block per (batch, chunk).
//   S = r * k^T with shifted keys.  Key row for e==0 of chunk 0 is out of
//   range: load a clamped (valid) row unconditionally and zero that S column
//   in the mask step -> no exec-mask divergence in the hot K loop.
// ---------------------------------------------------------------------------
__global__ void intra_kernel(const unsigned short* __restrict__ out_bf,
                             const unsigned short* __restrict__ v_bf,
                             float* __restrict__ reads,
                             const float* __restrict__ decay) {
  __shared__ unsigned short Smat[64][72];
  __shared__ unsigned short vTs[64][72];

  int b = blockIdx.x >> 6;         // NC_ = 64 chunks
  int n = blockIdx.x & 63;
  long base = (long)b * T_;
  long rowbase = base + (long)n * C_;

  int tid = threadIdx.x;
  int w = tid >> 5, lane = tid & 31, g = lane >> 4, hn = lane & 15;

  float gamma = 1.f / (1.f + expf(-decay[0]));
  float lg = logf(gamma);

  // ---- S = r * k^T  (wave w: rows 16w..16w+15, all 64 cols) ----
  v8f sc0 = v8zero(), sc1 = v8zero(), sc2 = v8zero(), sc3 = v8zero();
  const unsigned short* arow = out_bf + (rowbase + 16 * w + hn) * (size_t)D_;
  const unsigned short* krow[4];
#pragma unroll
  for (int nt = 0; nt < 4; nt++) {
    long e = nt * 16 + hn;
    long r = rowbase + e - 1;            // shifted key row (clamp if OOB)
    krow[nt] = out_bf + ((r >= base) ? r : base) * (size_t)D_;
  }
  for (int kt = 0; kt < D_ / 32; kt++) {
    int k0 = kt * 32;
    Frag a;
    load_a16(a, arow, k0, g);
    Frag b0, b1, b2, b3;
    load_b16(b0, krow[0], k0, g);
    load_b16(b1, krow[1], k0, g);
    load_b16(b2, krow[2], k0, g);
    load_b16(b3, krow[3], k0, g);
    sc0 = wmma_bf16(a.v, b0.v, sc0);
    sc1 = wmma_bf16(a.v, b1.v, sc1);
    sc2 = wmma_bf16(a.v, b2.v, sc2);
    sc3 = wmma_bf16(a.v, b3.v, sc3);
  }
  // mask S with decay matrix M[c,e] = gamma^(c-1-e) * (c>e); kill e==0 col of
  // chunk 0 (its key row is the zero pad).
#pragma unroll
  for (int nt = 0; nt < 4; nt++) {
    v8f acc = (nt == 0) ? sc0 : (nt == 1) ? sc1 : (nt == 2) ? sc2 : sc3;
#pragma unroll
    for (int v = 0; v < 8; v++) {
      int c = 16 * w + v + 8 * g;
      int e = nt * 16 + hn;
      bool live = (c > e) && !(n == 0 && e == 0);
      float m = live ? expf((float)(c - 1 - e) * lg) : 0.f;
      Smat[c][e] = f2bf(acc[v] * m);
    }
  }

  // ---- intra = (S o M) * v, 64-wide d segments ----
  for (int seg = 0; seg < D_ / 64; seg++) {
    __syncthreads();
    {  // stage v^T segment: vTs[d][e] = v[e, seg*64+d]
      int d = tid & 63, e0 = tid >> 6;
      for (int e = e0; e < 64; e += 2)
        vTs[d][e] = v_bf[(rowbase + e) * (size_t)D_ + seg * 64 + d];
    }
    __syncthreads();
    v8f ic0 = v8zero(), ic1 = v8zero(), ic2 = v8zero(), ic3 = v8zero();
#pragma unroll
    for (int s = 0; s < 2; s++) {
      Frag a;
      a.u[0] = *(const u32x4*)(&Smat[16 * w + hn][32 * s + 8 * g]);
      a.u[1] = *(const u32x4*)(&Smat[16 * w + hn][32 * s + 16 + 8 * g]);
#pragma unroll
      for (int nt = 0; nt < 4; nt++) {
        Frag bq;
        bq.u[0] = *(const u32x4*)(&vTs[nt * 16 + hn][32 * s + 8 * g]);
        bq.u[1] = *(const u32x4*)(&vTs[nt * 16 + hn][32 * s + 16 + 8 * g]);
        if (nt == 0) ic0 = wmma_bf16(a.v, bq.v, ic0);
        else if (nt == 1) ic1 = wmma_bf16(a.v, bq.v, ic1);
        else if (nt == 2) ic2 = wmma_bf16(a.v, bq.v, ic2);
        else ic3 = wmma_bf16(a.v, bq.v, ic3);
      }
    }
#pragma unroll
    for (int nt = 0; nt < 4; nt++) {
      v8f acc = (nt == 0) ? ic0 : (nt == 1) ? ic1 : (nt == 2) ? ic2 : ic3;
#pragma unroll
      for (int v = 0; v < 8; v++) {
        int c = 16 * w + v + 8 * g;
        int dl = nt * 16 + hn;
        reads[(rowbase + c) * (size_t)D_ + seg * 64 + dl] = acc[v];
      }
    }
  }
}

// ---------------------------------------------------------------------------
// K5: sequential chunk scan.  One 128-thread (4-wave) block per
// (batch, 16-row slice of W).
//   fp32 W state lives in REGISTERS: wave w owns, per 64-wide kd segment,
//   the 16x16 tile at cols seg*64+16w..+15, in C-fragment layout
//   (12 x v8f = 96 VGPRs/lane).  A bf16 shadow of W is kept in LDS and is
//   what `inter` consumes as B-fragments (pure ds_load_b128, no converts).
//   inter[c, vd] = gamma^c * sum_kd r[c,kd] * W[vd,kd]      (added to reads)
//   W[vd,kd]     = gamma^C * W + sum_c (v[c,vd]*gw[c]) * k[c,kd]
// ---------------------------------------------------------------------------
__global__ void scan_kernel(const unsigned short* __restrict__ out_bf,
                            const unsigned short* __restrict__ v_bf,
                            float* __restrict__ reads,
                            const float* __restrict__ decay) {
  constexpr int NJB = D_ / 16;  // 48
  constexpr int NSEG = D_ / 64; // 12
  __shared__ unsigned short Wbf[16][776];   // bf16 shadow of W slice
  __shared__ unsigned short vT[16][72];
  __shared__ unsigned short kT[64][72];

  int b = blockIdx.x / NJB;
  int jb = blockIdx.x % NJB;
  int vd0 = jb * 16;
  long base = (long)b * T_;

  int tid = threadIdx.x;
  int w = tid >> 5, lane = tid & 31, g = lane >> 4, hn = lane & 15;

  float gamma = 1.f / (1.f + expf(-decay[0]));
  float lg = logf(gamma);
  float gC = expf((float)C_ * lg);

  // register-resident fp32 W state
  v8f Wreg[NSEG];
#pragma unroll
  for (int s = 0; s < NSEG; s++) Wreg[s] = v8zero();
  // zero the bf16 shadow
  for (int i = tid; i < 16 * 776; i += 128) (&Wbf[0][0])[i] = 0;

  for (int n = 0; n < NC_; n++) {
    __syncthreads();  // W shadow writes of previous chunk complete
    long rowbase = base + (long)n * C_;

    // ---- inter: wave w handles c rows 16w..16w+15, cols = vd slice ----
    v8f acc = v8zero();
    const unsigned short* arow = out_bf + (rowbase + 16 * w + hn) * (size_t)D_;
    for (int kt = 0; kt < D_ / 32; kt++) {
      int k0 = kt * 32;
      Frag a, bw;
      load_a16(a, arow, k0, g);
      bw.u[0] = *(const u32x4*)(&Wbf[hn][k0 + 8 * g]);
      bw.u[1] = *(const u32x4*)(&Wbf[hn][k0 + 16 + 8 * g]);
      acc = wmma_bf16(a.v, bw.v, acc);
    }
#pragma unroll
    for (int v = 0; v < 8; v++) {
      int cc = 16 * w + v + 8 * g;
      reads[(rowbase + cc) * (size_t)D_ + vd0 + hn] +=
          acc[v] * expf((float)cc * lg);
    }

    // ---- stage v~^T (scaled by gw[c] = gamma^(63-c)) ----
    for (int idx = tid; idx < 16 * 64; idx += 128) {
      int c = idx >> 4, vd = idx & 15;
      float vv = bf2f(v_bf[(rowbase + c) * (size_t)D_ + vd0 + vd]) *
                 expf((float)(C_ - 1 - c) * lg);
      vT[vd][c] = f2bf(vv);
    }

    // ---- W update in 64-wide kd segments ----
#pragma unroll
    for (int seg = 0; seg < NSEG; seg++) {
      int kd0 = seg * 64;
      __syncthreads();  // prior segment compute done; (seg 0) inter done
      {  // stage k^T segment with causal shift: kT[kd][c] = k[c, kd0+kd]
        int kd = tid & 63, c0 = tid >> 6;
        for (int c = c0; c < 64; c += 2) {
          long gr = rowbase + c - 1;
          kT[kd][c] = (gr >= base)
                          ? out_bf[gr * (size_t)D_ + kd0 + kd]
                          : (unsigned short)0;
        }
      }
      __syncthreads();
      v8f u = v8zero();
#pragma unroll
      for (int s = 0; s < 2; s++) {
        Frag a, bq;
        a.u[0] = *(const u32x4*)(&vT[hn][32 * s + 8 * g]);
        a.u[1] = *(const u32x4*)(&vT[hn][32 * s + 16 + 8 * g]);
        bq.u[0] = *(const u32x4*)(&kT[16 * w + hn][32 * s + 8 * g]);
        bq.u[1] = *(const u32x4*)(&kT[16 * w + hn][32 * s + 16 + 8 * g]);
        u = wmma_bf16(a.v, bq.v, u);
      }
      // decay + accumulate in registers, refresh bf16 shadow
      int col = kd0 + 16 * w + hn;
#pragma unroll
      for (int v = 0; v < 8; v++) {
        float nw = gC * Wreg[seg][v] + u[v];
        Wreg[seg][v] = nw;
        Wbf[v + 8 * g][col] = f2bf(nw);
      }
    }
  }
}

// ---------------------------------------------------------------------------
// Host launcher
// ---------------------------------------------------------------------------
extern "C" void kernel_launch(void* const* d_in, const int* in_sizes, int n_in,
                              void* d_out, int out_size, void* d_ws,
                              size_t ws_size, hipStream_t stream) {
  const float* x       = (const float*)d_in[0];
  const float* norm_w  = (const float*)d_in[1];
  const float* proj_w  = (const float*)d_in[2];
  const float* gate_w  = (const float*)d_in[3];
  const float* conv_w  = (const float*)d_in[4];
  const float* conv_b  = (const float*)d_in[5];
  const float* outp_w  = (const float*)d_in[6];
  const float* write_w = (const float*)d_in[7];
  const float* read_w  = (const float*)d_in[8];
  const float* decay   = (const float*)d_in[9];
  const float* logalph = (const float*)d_in[10];

  unsigned char* ws = (unsigned char*)d_ws;
  size_t off = 0;
  auto alloc = [&](size_t bytes) {
    size_t o = off;
    off = (off + bytes + 255) & ~(size_t)255;
    return o;
  };
  unsigned short* proj_bf  = (unsigned short*)(ws + alloc((size_t)DI_ * D_ * 2));
  unsigned short* gate_bf  = (unsigned short*)(ws + alloc((size_t)DI_ * D_ * 2));
  unsigned short* outp_bf  = (unsigned short*)(ws + alloc((size_t)D_ * DI_ * 2));
  unsigned short* write_bf = (unsigned short*)(ws + alloc((size_t)D_ * D_ * 2));
  unsigned short* read_bf  = (unsigned short*)(ws + alloc((size_t)D_ * D_ * 2));
  unsigned short* xn_bf    = (unsigned short*)(ws + alloc((size_t)BT_ * D_ * 2));
  unsigned short* val_bf   = (unsigned short*)(ws + alloc((size_t)BT_ * DI_ * 2));
  unsigned short* gact_bf  = (unsigned short*)(ws + alloc((size_t)BT_ * DI_ * 2));
  float*          out_f    = (float*)(ws + alloc((size_t)BT_ * D_ * 4));
  unsigned short* out_bf   = (unsigned short*)(ws + alloc((size_t)BT_ * D_ * 2));
  unsigned short* v_bf     = (unsigned short*)(ws + alloc((size_t)BT_ * D_ * 2));
  float*          reads_f  = (float*)(ws + alloc((size_t)BT_ * D_ * 4));
  (void)ws_size; (void)n_in; (void)in_sizes; (void)out_size;

  // weight casts
  cast_kernel<<<(DI_ * D_ + 255) / 256, 256, 0, stream>>>(proj_w, proj_bf, DI_ * D_);
  cast_kernel<<<(DI_ * D_ + 255) / 256, 256, 0, stream>>>(gate_w, gate_bf, DI_ * D_);
  cast_kernel<<<(D_ * DI_ + 255) / 256, 256, 0, stream>>>(outp_w, outp_bf, D_ * DI_);
  cast_kernel<<<(D_ * D_ + 255) / 256, 256, 0, stream>>>(write_w, write_bf, D_ * D_);
  cast_kernel<<<(D_ * D_ + 255) / 256, 256, 0, stream>>>(read_w, read_bf, D_ * D_);

  // rmsnorm + cast
  rmsnorm_kernel<<<BT_, 256, 0, stream>>>(x, norm_w, xn_bf);

  // projections
  gemm_kernel<0, false><<<dim3(BT_ / 128, DI_ / 64), 256, 0, stream>>>(
      xn_bf, proj_bf, BT_, DI_, D_, nullptr, val_bf, nullptr, nullptr, nullptr);
  gemm_kernel<1, false><<<dim3(BT_ / 128, DI_ / 64), 256, 0, stream>>>(
      xn_bf, gate_bf, BT_, DI_, D_, nullptr, gact_bf, nullptr, nullptr, nullptr);

  // depthwise causal conv + silu, * silu(gate)  (in-place into gact_bf)
  conv_silu_gate_kernel<<<(size_t)BT_ * DI_ / 256, 256, 0, stream>>>(
      val_bf, gact_bf, conv_w, conv_b);

  // out projection (fp32 + bf16)
  gemm_kernel<2, false><<<dim3(BT_ / 128, D_ / 64), 256, 0, stream>>>(
      gact_bf, outp_bf, BT_, D_, DI_, out_f, out_bf, nullptr, nullptr, nullptr);

  // write projection -> v
  gemm_kernel<0, false><<<dim3(BT_ / 128, D_ / 64), 256, 0, stream>>>(
      out_bf, write_bf, BT_, D_, D_, nullptr, v_bf, nullptr, nullptr, nullptr);

  // intra-chunk attention (initializes reads)
  intra_kernel<<<B_ * NC_, 128, 0, stream>>>(out_bf, v_bf, reads_f, decay);

  // sequential memory scan (adds inter into reads)
  scan_kernel<<<B_ * (D_ / 16), 128, 0, stream>>>(out_bf, v_bf, reads_f, decay);

  // read projection + fused residual:  d_out = x + out + exp(log_alpha)*read_out
  gemm_kernel<3, true><<<dim3(BT_ / 128, D_ / 64), 256, 0, stream>>>(
      reads_f, read_bf, BT_, D_, D_, (float*)d_out, nullptr, x, out_f, logalph);
}
